// ReachabilityGNN_13108240187815
// MI455X (gfx1250) — matrix operand order repaired and checked by reading, not compiled, not run on hardware
//
#include <hip/hip_runtime.h>
#include <hip/hip_bf16.h>

typedef __attribute__((ext_vector_type(16))) _Float16 v16h;
typedef __attribute__((ext_vector_type(8)))  _Float16 v8h;
typedef __attribute__((ext_vector_type(8)))  float    v8f;

#define NODES   100000
#define FEAT_IN 128
#define HID     64
#define NOUT    4

// ---------------- utility conversion kernels ----------------

__global__ void k_cvt_f32_to_f16(const float* __restrict__ in,
                                 _Float16* __restrict__ out, int n) {
    int t = blockIdx.x * blockDim.x + threadIdx.x;
    if (t < n) out[t] = (_Float16)in[t];
}

// W[K x Ncol] row-major f32 -> Wt[Ncol x K] (column-major of W) f16
__global__ void k_wt_cvt(const float* __restrict__ W,
                         _Float16* __restrict__ Wt, int K, int Ncol) {
    int t = blockIdx.x * blockDim.x + threadIdx.x;
    if (t >= K * Ncol) return;
    int k = t / Ncol, n = t % Ncol;
    Wt[(size_t)n * K + k] = (_Float16)W[t];
}

// ---------------- degree / normalization ----------------

__global__ void k_deg_init(float* __restrict__ deg, int n) {
    int t = blockIdx.x * blockDim.x + threadIdx.x;
    if (t < n) deg[t] = 1.0f;   // self-loop contributes 1 to in-degree
}

__global__ void k_deg_acc(const int* __restrict__ dst, float* __restrict__ deg, int e) {
    int t = blockIdx.x * blockDim.x + threadIdx.x;
    if (t < e) atomicAdd(&deg[dst[t]], 1.0f);
}

__global__ void k_deg_isqrt(float* __restrict__ deg, int n) {
    int t = blockIdx.x * blockDim.x + threadIdx.x;
    if (t < n) {
        float d = deg[t];
        deg[t] = (d > 0.0f) ? rsqrtf(d) : 0.0f;   // in-place -> dinv
    }
}

// ---- WMMA GEMM: one wave computes a full 16x64 stripe (4 accumulators) ----
// C[M x 64] = A[M x K](f16, row-major) * W[K x 64] (given as Wt[64 x K] f16)

__global__ __launch_bounds__(256)
void k_gemm_wmma(const _Float16* __restrict__ A,
                 const _Float16* __restrict__ Bt,   // [64 x K] col-major of W
                 float* __restrict__ C,
                 int M, int K) {
    int wave   = blockIdx.x * (blockDim.x >> 5) + (threadIdx.x >> 5);
    int tilesM = M >> 4;
    if (wave >= tilesM) return;                    // uniform per wave: whole wave exits

    int lane = threadIdx.x & 31;
    int hi   = lane >> 4;                          // 0 | 1
    int lo   = lane & 15;

    const _Float16* arow  = A  + (size_t)(wave * 16 + lo) * K;  // A row for this lane
    const _Float16* bbase = Bt + (size_t)lo * K;                // col lo of output tile 0

    v8f acc0 = {}, acc1 = {}, acc2 = {}, acc3 = {};

    for (int k0 = 0; k0 < K; k0 += 32) {
        // A fragment (contiguous k order, matches B): VGPR0-3 = K[k0+16*hi..+7], VGPR4-7 = K[k0+16*hi+8..+15]
        union { v16h v; v8h h[2]; } a;
        a.h[0] = *(const v8h*)(arow + k0 + hi * 16);
        a.h[1] = *(const v8h*)(arow + k0 + hi * 16 + 8);

        // B fragments for the 4 column tiles (lanes 0-15: K=k0..k0+15; 16-31: +16)
        const _Float16* bp = bbase + k0 + hi * 16;
        union { v16h v; v8h h[2]; } b0, b1, b2, b3;
        b0.h[0] = *(const v8h*)(bp);                b0.h[1] = *(const v8h*)(bp + 8);
        b1.h[0] = *(const v8h*)(bp + 16 * K);       b1.h[1] = *(const v8h*)(bp + 16 * K + 8);
        b2.h[0] = *(const v8h*)(bp + 32 * K);       b2.h[1] = *(const v8h*)(bp + 32 * K + 8);
        b3.h[0] = *(const v8h*)(bp + 48 * K);       b3.h[1] = *(const v8h*)(bp + 48 * K + 8);

        acc0 = __builtin_amdgcn_wmma_f32_16x16x32_f16(false, a.v, false, b0.v, (short)0, acc0, false, false);
        acc1 = __builtin_amdgcn_wmma_f32_16x16x32_f16(false, a.v, false, b1.v, (short)0, acc1, false, false);
        acc2 = __builtin_amdgcn_wmma_f32_16x16x32_f16(false, a.v, false, b2.v, (short)0, acc2, false, false);
        acc3 = __builtin_amdgcn_wmma_f32_16x16x32_f16(false, a.v, false, b3.v, (short)0, acc3, false, false);
    }

    // C/D layout: VGPR j holds (row = j + 8*hi, col = lo) within each 16x16 tile
    float* crow = C + (size_t)(wave * 16 + hi * 8) * HID + lo;
#pragma unroll
    for (int j = 0; j < 8; ++j) {
        crow[(size_t)j * HID +  0] = acc0[j];
        crow[(size_t)j * HID + 16] = acc1[j];
        crow[(size_t)j * HID + 32] = acc2[j];
        crow[(size_t)j * HID + 48] = acc3[j];
    }
}

// ---------------- graph aggregation ----------------

// agg[i][:] = h[i][:] * dinv[i]^2      (self-loop term, also initializes agg)
__global__ void k_agg_self(const float* __restrict__ h,
                           const float* __restrict__ dinv,
                           float* __restrict__ agg, int n) {
    int t = blockIdx.x * blockDim.x + threadIdx.x;
    if (t >= n * HID) return;
    float di = dinv[t >> 6];
    agg[t] = h[t] * di * di;
}

// 16 lanes per edge, float4 per lane: agg[dst] += h[src] * dinv[src]*dinv[dst]
// Gathers hit L2 (25.6 MB feature table << 192 MB); scatter is f32 L2 atomics.
__global__ __launch_bounds__(256)
void k_edge_msg(const int* __restrict__ src, const int* __restrict__ dst,
                const float* __restrict__ dinv,
                const float* __restrict__ h, float* __restrict__ agg, int e) {
    int t = blockIdx.x * blockDim.x + threadIdx.x;
    int edge = t >> 4;
    if (edge >= e) return;
    int lane = t & 15;

    // prefetch upcoming index cache lines (speculative, lowers to global_prefetch_b8)
    __builtin_prefetch(&src[edge + 2048], 0, 0);
    __builtin_prefetch(&dst[edge + 2048], 0, 0);

    int s = src[edge], d = dst[edge];
    float w = dinv[s] * dinv[d];
    const float4 v = ((const float4*)(h + (size_t)s * HID))[lane];
    float* ap = agg + (size_t)d * HID + lane * 4;
    atomicAdd(ap + 0, v.x * w);
    atomicAdd(ap + 1, v.y * w);
    atomicAdd(ap + 2, v.z * w);
    atomicAdd(ap + 3, v.w * w);
}

// ---------------- bias + ReLU + LayerNorm, f16 output (one node per wave32) ----

__global__ __launch_bounds__(256)
void k_post_ln(const float* __restrict__ agg, const float* __restrict__ bias,
               const float* __restrict__ gamma, const float* __restrict__ beta,
               _Float16* __restrict__ out, int n) {
    int t = blockIdx.x * blockDim.x + threadIdx.x;
    int node = t >> 5;
    if (node >= n) return;
    int lane = t & 31;
    int c = lane * 2;

    float x0 = fmaxf(agg[(size_t)node * HID + c + 0] + bias[c + 0], 0.0f);
    float x1 = fmaxf(agg[(size_t)node * HID + c + 1] + bias[c + 1], 0.0f);

    float s = x0 + x1;
    for (int m = 16; m >= 1; m >>= 1) s += __shfl_xor(s, m, 32);
    float mu = s * (1.0f / 64.0f);

    float d0 = x0 - mu, d1 = x1 - mu;
    float q = d0 * d0 + d1 * d1;
    for (int m = 16; m >= 1; m >>= 1) q += __shfl_xor(q, m, 32);
    float rstd = rsqrtf(q * (1.0f / 64.0f) + 1e-5f);

    out[(size_t)node * HID + c + 0] = (_Float16)(d0 * rstd * gamma[c + 0] + beta[c + 0]);
    out[(size_t)node * HID + c + 1] = (_Float16)(d1 * rstd * gamma[c + 1] + beta[c + 1]);
}

// ---------------- classifier: out[N x 4] = h(f16) @ Wc + bc ----------------

__global__ void k_classifier(const _Float16* __restrict__ h,
                             const float* __restrict__ Wc,
                             const float* __restrict__ bc,
                             float* __restrict__ out, int n) {
    int node = blockIdx.x * blockDim.x + threadIdx.x;
    if (node >= n) return;
    float a0 = bc[0], a1 = bc[1], a2 = bc[2], a3 = bc[3];
    const _Float16* hp = h + (size_t)node * HID;
#pragma unroll 8
    for (int k = 0; k < HID; ++k) {
        float x = (float)hp[k];
        a0 += x * Wc[k * 4 + 0];
        a1 += x * Wc[k * 4 + 1];
        a2 += x * Wc[k * 4 + 2];
        a3 += x * Wc[k * 4 + 3];
    }
    float* op = out + (size_t)node * 4;
    op[0] = a0; op[1] = a1; op[2] = a2; op[3] = a3;
}

// ---------------- orchestration ----------------

static inline size_t align256(size_t x) { return (x + 255) & ~(size_t)255; }

extern "C" void kernel_launch(void* const* d_in, const int* in_sizes, int n_in,
                              void* d_out, int out_size, void* d_ws, size_t ws_size,
                              hipStream_t stream) {
    const float* x     = (const float*)d_in[0];
    const int*   ei    = (const int*)d_in[1];
    const float* W1    = (const float*)d_in[2];
    const float* b1    = (const float*)d_in[3];
    const float* W2    = (const float*)d_in[4];
    const float* b2    = (const float*)d_in[5];
    const float* W3    = (const float*)d_in[6];
    const float* b3    = (const float*)d_in[7];
    const float* gamma = (const float*)d_in[8];
    const float* beta  = (const float*)d_in[9];
    const float* Wc    = (const float*)d_in[10];
    const float* bc    = (const float*)d_in[11];
    float* out = (float*)d_out;

    const int N = in_sizes[0] / FEAT_IN;   // 100000
    const int E = in_sizes[1] / 2;         // 3200000
    const int* src = ei;
    const int* dst = ei + E;

    // workspace carve-up
    char* p = (char*)d_ws;
    float*    deg   = (float*)p;    p += align256((size_t)N * sizeof(float));       // -> dinv in-place
    _Float16* Xh    = (_Float16*)p; p += align256((size_t)N * FEAT_IN * 2);
    _Float16* Hh    = (_Float16*)p; p += align256((size_t)N * HID * 2);
    float*    Hg    = (float*)p;    p += align256((size_t)N * HID * sizeof(float));
    float*    agg   = (float*)p;    p += align256((size_t)N * HID * sizeof(float));
    _Float16* W1t   = (_Float16*)p; p += align256((size_t)FEAT_IN * HID * 2);
    _Float16* W2t   = (_Float16*)p; p += align256((size_t)HID * HID * 2);
    _Float16* W3t   = (_Float16*)p; p += align256((size_t)HID * HID * 2);
    (void)ws_size; (void)n_in; (void)out_size;

    const int T = 256;
    // 1) normalization coefficients
    k_deg_init <<<(N + T - 1) / T, T, 0, stream>>>(deg, N);
    k_deg_acc  <<<(E + T - 1) / T, T, 0, stream>>>(dst, deg, E);
    k_deg_isqrt<<<(N + T - 1) / T, T, 0, stream>>>(deg, N);

    // 2) precision conversions (X and transposed weights -> f16)
    k_cvt_f32_to_f16<<<((N * FEAT_IN) + T - 1) / T, T, 0, stream>>>(x, Xh, N * FEAT_IN);
    k_wt_cvt<<<((FEAT_IN * HID) + T - 1) / T, T, 0, stream>>>(W1, W1t, FEAT_IN, HID);
    k_wt_cvt<<<((HID * HID) + T - 1) / T, T, 0, stream>>>(W2, W2t, HID, HID);
    k_wt_cvt<<<((HID * HID) + T - 1) / T, T, 0, stream>>>(W3, W3t, HID, HID);

    const int wavesPerBlock = T / 32;                       // 8 waves/block
    const int stripes = N / 16;                             // 6250 row-stripes, 16x64 each
    const int gemmBlocks = (stripes + wavesPerBlock - 1) / wavesPerBlock;
    const int elemBlocks = ((N * HID) + T - 1) / T;
    const int edgeBlocks = ((E * 16) + T - 1) / T;          // 16 lanes per edge
    const int nodeBlocks = ((N * 32) + T - 1) / T;

    const _Float16* Wts[3]    = { W1t, W2t, W3t };
    const float*    biases[3] = { b1, b2, b3 };

    for (int layer = 0; layer < 3; ++layer) {
        const _Float16* Ain = (layer == 0) ? Xh : Hh;
        const int K = (layer == 0) ? FEAT_IN : HID;

        k_gemm_wmma<<<gemmBlocks, T, 0, stream>>>(Ain, Wts[layer], Hg, N, K);
        k_agg_self <<<elemBlocks, T, 0, stream>>>(Hg, deg, agg, N);
        k_edge_msg <<<edgeBlocks, T, 0, stream>>>(src, dst, deg, Hg, agg, E);
        k_post_ln  <<<nodeBlocks, T, 0, stream>>>(agg, biases[layer], gamma, beta, Hh, N);
    }

    k_classifier<<<(N + T - 1) / T, T, 0, stream>>>(Hh, Wc, bc, out, N);
}
